// MambaLayer_71966472012605
// MI455X (gfx1250) — matrix operand order repaired and checked by reading, not compiled, not run on hardware
//
#include <hip/hip_runtime.h>

typedef __bf16 bf16;
typedef bf16  v16bf __attribute__((ext_vector_type(16)));
typedef float v8f   __attribute__((ext_vector_type(8)));

#define BATCH 2
#define CDIM  128
#define LSEQ  8192
#define LSH   13
#define LMASK (LSEQ - 1)
#define DI    256      // d_inner
#define E2    512      // 2*d_inner
#define NST   16       // d_state
#define RNK   8        // dt_rank
#define BL    (BATCH * LSEQ)

// ---------------------------------------------------------------------------
// Pack a row-major f32 (M x K) weight into CDNA5 WMMA A-fragment layout:
//   dst[(mt*Ktiles+kt)*512 + lane*16 + j]  (bf16), 32B contiguous per lane.
// A-matrix lane map used here (contiguous K): lane<16 -> K {0..15},
// lane>=16 -> K {16..31} within the 32-wide K tile.
// flipK reverses the K index (implements channel-flipped mamba direction).
// ---------------------------------------------------------------------------
__global__ void pack_a_kernel(const float* __restrict__ src, bf16* __restrict__ dst,
                              int Mreal, int Mtiles, int K, int flipK)
{
    int tid = blockIdx.x * 256 + threadIdx.x;
    int Ktiles = K >> 5;
    int total = Mtiles * Ktiles * 512;
    if (tid >= total) return;
    int frag = tid >> 9;
    int r    = tid & 511;
    int lane = r >> 4;
    int j    = r & 15;
    int kt   = frag % Ktiles;
    int mt   = frag / Ktiles;
    int m    = mt * 16 + (lane & 15);
    int half = lane >> 4;
    int vg = j >> 1, sub = j & 1;
    int k = kt * 32 + 16 * half + j;
    float v = 0.f;
    if (m < Mreal) {
        int kk = flipK ? (K - 1 - k) : k;
        v = src[(size_t)m * K + kk];
    }
    dst[tid] = (bf16)v;
}

// ---------------------------------------------------------------------------
// Generic WMMA GEMM: out[b][m][l] = sum_k Apack[m][k] * Bact[b][lr][k]
//   Apack: pre-packed A fragments (Mtiles x Ktiles x 32 lanes x 16 bf16)
//   Bact : bf16 activations, [b][l][K] with K contiguous (B-fragment = one
//          32B load per lane: lane half selects K 0..15 / 16..31 of the tile)
//   flipL: read B rows sequence-reversed (mamba direction handling)
// One wave per 16x16 D tile; 4 waves / 128-thread block.
// ---------------------------------------------------------------------------
__global__ void wmma_gemm_kernel(const bf16* __restrict__ Apack,
                                 const bf16* __restrict__ Bact,
                                 float* __restrict__ out,
                                 int Mtiles, int Ktiles, int Mrows, int flipL)
{
    int lane  = threadIdx.x & 31;
    int wv    = threadIdx.x >> 5;
    int mtile = blockIdx.y * 4 + wv;
    if (mtile >= Mtiles) return;            // uniform per wave
    int ntile = blockIdx.x;
    int n = lane & 15, half = lane >> 4;
    int p = ntile * 16 + n;                 // global position (b,l)
    int b = p >> LSH;
    int l = p & LMASK;
    int lr = flipL ? (LSEQ - 1 - l) : l;
    int K = Ktiles << 5;

    const v16bf* Afr  = (const v16bf*)Apack;
    const bf16*  Brow = Bact + ((size_t)b * LSEQ + lr) * K + half * 16;

    v8f acc = {};
    for (int kt = 0; kt < Ktiles; ++kt) {
        v16bf a  = Afr[((size_t)mtile * Ktiles + kt) * 32 + lane];
        v16bf bb = *(const v16bf*)(Brow + kt * 32);
        acc = __builtin_amdgcn_wmma_f32_16x16x32_bf16(
            false, a, false, bb, (short)0, acc, false, false);
    }
    // D layout: lanes 0-15 rows 0..7, lanes 16-31 rows 8..15; col = lane&15
    float* orow = out + ((size_t)b * Mrows + (size_t)mtile * 16 + half * 8) * LSEQ + l;
#pragma unroll
    for (int r8 = 0; r8 < 8; ++r8)
        orow[(size_t)r8 * LSEQ] = acc[r8];
}

// ---------------------------------------------------------------------------
// out_proj GEMM with direction-unflip accumulate into x_mamba[b][l][o] (f32).
// ---------------------------------------------------------------------------
__global__ void wmma_outproj_kernel(const bf16* __restrict__ Apack,
                                    const bf16* __restrict__ Bact,
                                    float* __restrict__ xmamba,
                                    int flipL, int flipC)
{
    const int Ktiles = DI / 32;             // 8
    int lane  = threadIdx.x & 31;
    int wv    = threadIdx.x >> 5;
    int mtile = blockIdx.y * 4 + wv;        // 0..7 (M=128)
    int ntile = blockIdx.x;
    int n = lane & 15, half = lane >> 4;
    int p = ntile * 16 + n;
    int b = p >> LSH;
    int l = p & LMASK;

    const bf16* Brow = Bact + (size_t)p * DI + half * 16;
    v8f acc = {};
    for (int kt = 0; kt < Ktiles; ++kt) {
        v16bf a  = ((const v16bf*)Apack)[((size_t)mtile * Ktiles + kt) * 32 + lane];
        v16bf bb = *(const v16bf*)(Brow + kt * 32);
        acc = __builtin_amdgcn_wmma_f32_16x16x32_bf16(
            false, a, false, bb, (short)0, acc, false, false);
    }
    int lout = flipL ? (LSEQ - 1 - l) : l;
    float* dst = xmamba + ((size_t)b * LSEQ + lout) * CDIM;
#pragma unroll
    for (int r8 = 0; r8 < 8; ++r8) {
        int m = mtile * 16 + half * 8 + r8;
        int o = flipC ? (CDIM - 1 - m) : m;
        dst[o] += acc[r8];                  // distinct (o,l) per lane-elem: no race
    }
}

// --------------------------- elementwise kernels ---------------------------

__global__ void transpose_x_kernel(const float* __restrict__ x, bf16* __restrict__ xb)
{
    size_t i = (size_t)blockIdx.x * 256 + threadIdx.x;     // over B*C*L
    if (i >= (size_t)BATCH * CDIM * LSEQ) return;
    int l = (int)(i & LMASK);
    size_t bc = i >> LSH;
    int c = (int)(bc % CDIM);
    int b = (int)(bc / CDIM);
    xb[((size_t)b * LSEQ + l) * CDIM + c] = (bf16)x[i];
}

__global__ void bn_stats_kernel(const float* __restrict__ pre,
                                const float* __restrict__ g,
                                const float* __restrict__ be,
                                float* __restrict__ scale,
                                float* __restrict__ shift)
{
    __shared__ float s1[256], s2[256];
    int c = blockIdx.x;
    float a = 0.f, q = 0.f;
    for (int j = threadIdx.x; j < BATCH * LSEQ; j += 256) {
        int b = j >> LSH;
        int l = j & LMASK;
        float v = pre[((size_t)b * CDIM + c) * LSEQ + l];
        a += v; q += v * v;
    }
    s1[threadIdx.x] = a; s2[threadIdx.x] = q;
    __syncthreads();
    for (int st = 128; st > 0; st >>= 1) {
        if (threadIdx.x < st) {
            s1[threadIdx.x] += s1[threadIdx.x + st];
            s2[threadIdx.x] += s2[threadIdx.x + st];
        }
        __syncthreads();
    }
    if (threadIdx.x == 0) {
        float inv  = 1.f / (float)(BATCH * LSEQ);
        float mean = s1[0] * inv;
        float var  = s2[0] * inv - mean * mean;
        float sc   = g[c] * rsqrtf(var + 1e-5f);
        scale[c] = sc;
        shift[c] = be[c] - mean * sc;
    }
}

__global__ void bn_relu_t_kernel(const float* __restrict__ pre,
                                 const float* __restrict__ scale,
                                 const float* __restrict__ shift,
                                 float* __restrict__ xnf, bf16* __restrict__ xnb)
{
    size_t i = (size_t)blockIdx.x * 256 + threadIdx.x;     // over B*C*L, [b][c][l]
    if (i >= (size_t)BATCH * CDIM * LSEQ) return;
    int l = (int)(i & LMASK);
    size_t bc = i >> LSH;
    int c = (int)(bc % CDIM);
    int b = (int)(bc / CDIM);
    float v = fmaxf(pre[i] * scale[c] + shift[c], 0.f);
    size_t o = ((size_t)b * LSEQ + l) * CDIM + c;
    xnf[o] = v;
    xnb[o] = (bf16)v;
}

__global__ void bn_relu_out_kernel(const float* __restrict__ pre,
                                   const float* __restrict__ scale,
                                   const float* __restrict__ shift,
                                   float* __restrict__ out)
{
    size_t i = (size_t)blockIdx.x * 256 + threadIdx.x;     // [b][o][l]
    if (i >= (size_t)BATCH * CDIM * LSEQ) return;
    int c = (int)((i >> LSH) % CDIM);
    out[i] = fmaxf(pre[i] * scale[c] + shift[c], 0.f);
}

__global__ void conv_silu_kernel(const float* __restrict__ xz,
                                 const float* __restrict__ conv_w,
                                 const float* __restrict__ conv_b,
                                 float* __restrict__ xca, bf16* __restrict__ xcab)
{
    size_t i = (size_t)blockIdx.x * 256 + threadIdx.x;     // over B*DI*L
    if (i >= (size_t)BATCH * DI * LSEQ) return;
    int l = (int)(i & LMASK);
    size_t bd = i >> LSH;
    int d = (int)(bd % DI);
    int b = (int)(bd / DI);
    const float* src = xz + ((size_t)b * E2 + d) * LSEQ;   // first E2/2 rows = xc
    const float* w = conv_w + d * 4;
    float acc = conv_b[d];
#pragma unroll
    for (int t = 0; t < 4; ++t) {
        int ls = l - 3 + t;
        if (ls >= 0) acc += w[t] * src[ls];
    }
    float s = acc / (1.f + __expf(-acc));                  // silu
    xca[i] = s;
    xcab[((size_t)b * LSEQ + l) * DI + d] = (bf16)s;
}

__global__ void dt_softplus_kernel(const float* __restrict__ xdbl,
                                   const float* __restrict__ w,
                                   const float* __restrict__ bias,
                                   float* __restrict__ delta)
{
    size_t i = (size_t)blockIdx.x * 256 + threadIdx.x;     // over B*DI*L
    if (i >= (size_t)BATCH * DI * LSEQ) return;
    int l = (int)(i & LMASK);
    size_t bd = i >> LSH;
    int d = (int)(bd % DI);
    int b = (int)(bd / DI);
    float s = bias[d];
    const float* base = xdbl + (size_t)b * 48 * LSEQ + l;  // rows 0..7 = dt_lo
#pragma unroll
    for (int r = 0; r < RNK; ++r)
        s += w[d * RNK + r] * base[(size_t)r * LSEQ];
    delta[i] = (s > 20.f) ? s : log1pf(__expf(s));
}

// ---------------------------------------------------------------------------
// Selective scan: one wave32 per two d-channels; 16 lanes = n state dim.
//   h = exp(delta*A)*h + (delta*u)*B ;  y = sum_n h*C ; y = (y + Dp*u)*silu(z)
// n-reduction via 4 xor-shuffles inside the 16-lane half.
// ---------------------------------------------------------------------------
__global__ void ssm_scan_kernel(const float* __restrict__ delta,
                                const float* __restrict__ u,
                                const float* __restrict__ xdbl,
                                const float* __restrict__ xz,
                                const float* __restrict__ A_log,
                                const float* __restrict__ Dp,
                                bf16* __restrict__ ybf)
{
    int lane = threadIdx.x & 31;
    int n = lane & 15;
    int idx = blockIdx.x;                                  // 0 .. BATCH*128-1
    int b = idx >> 7;
    int d = ((idx & 127) << 1) + (lane >> 4);
    const float* dP  = delta + ((size_t)b * DI + d) * LSEQ;
    const float* uP  = u     + ((size_t)b * DI + d) * LSEQ;
    const float* zP  = xz    + ((size_t)b * E2 + DI + d) * LSEQ;
    const float* BmP = xdbl  + ((size_t)b * 48 + RNK + n) * LSEQ;
    const float* CmP = xdbl  + ((size_t)b * 48 + RNK + NST + n) * LSEQ;
    float Ac = -__expf(A_log[d * NST + n]);
    float Dd = Dp[d];
    bf16* yO = ybf + (size_t)b * LSEQ * DI + d;
    float h = 0.f;
    for (int t = 0; t < LSEQ; ++t) {
        float dl = dP[t];
        float ut = uP[t];
        float bm = BmP[t];
        float cm = CmP[t];
        h = __expf(dl * Ac) * h + (dl * ut) * bm;
        float ps = h * cm;
        ps += __shfl_xor(ps, 1, 32);
        ps += __shfl_xor(ps, 2, 32);
        ps += __shfl_xor(ps, 4, 32);
        ps += __shfl_xor(ps, 8, 32);
        if (n == 0) {
            float zv  = zP[t];
            float y   = ps + Dd * ut;
            float sil = zv / (1.f + __expf(-zv));
            yO[(size_t)t * DI] = (bf16)(y * sil);
        }
    }
}

__global__ void add_avg_kernel(const float* __restrict__ xm,
                               const float* __restrict__ xnf,
                               bf16* __restrict__ s)
{
    size_t i = (size_t)blockIdx.x * 256 + threadIdx.x;     // over BL*CDIM, [b][l][c]
    if (i >= (size_t)BL * CDIM) return;
    s[i] = (bf16)(0.25f * xm[i] + xnf[i]);
}

// ---------------------------------------------------------------------------

static inline int blk256(size_t n) { return (int)((n + 255) / 256); }

extern "C" void kernel_launch(void* const* d_in, const int* in_sizes, int n_in,
                              void* d_out, int out_size, void* d_ws, size_t ws_size,
                              hipStream_t stream)
{
    (void)in_sizes; (void)n_in; (void)out_size; (void)ws_size;

    const float* x         = (const float*)d_in[0];
    const float* nin_w     = (const float*)d_in[1];
    const float* nin2_w    = (const float*)d_in[2];
    const float* gamma     = (const float*)d_in[3];
    const float* beta      = (const float*)d_in[4];
    const float* gamma2    = (const float*)d_in[5];
    const float* beta2     = (const float*)d_in[6];
    const float* in_proj_w = (const float*)d_in[7];
    const float* conv_w    = (const float*)d_in[8];
    const float* conv_b    = (const float*)d_in[9];
    const float* x_proj_w  = (const float*)d_in[10];
    const float* dt_proj_w = (const float*)d_in[11];
    const float* dt_proj_b = (const float*)d_in[12];
    const float* A_log     = (const float*)d_in[13];
    const float* Dp        = (const float*)d_in[14];
    const float* out_proj_w= (const float*)d_in[15];
    float* out = (float*)d_out;

    // ----- workspace carve-out (256B aligned) -----
    char* ws = (char*)d_ws;
    size_t off = 0;
    auto alloc = [&](size_t bytes) -> char* {
        char* p = ws + off;
        off = (off + bytes + 255) & ~(size_t)255;
        return p;
    };
    bf16*  xb16   = (bf16*) alloc((size_t)BL * CDIM * 2);   // x transposed -> [b][l][c]
    float* pre1   = (float*)alloc((size_t)BATCH * CDIM * LSEQ * 4); // nin out (reused for nin2)
    float* xn_f32 = (float*)alloc((size_t)BL * CDIM * 4);   // act_x / mamba input [b][l][c]
    bf16*  xn_b16 = (bf16*) alloc((size_t)BL * CDIM * 2);
    float* xmamba = (float*)alloc((size_t)BL * CDIM * 4);   // 4-direction accumulator
    float* xz     = (float*)alloc((size_t)BATCH * E2 * LSEQ * 4);  // in_proj out [b][512][l]
    float* xca    = (float*)alloc((size_t)BATCH * DI * LSEQ * 4);  // conv+silu [b][d][l]
    bf16*  xcab   = (bf16*) alloc((size_t)BL * DI * 2);            // [b][l][d]
    float* xdbl   = (float*)alloc((size_t)BATCH * 48 * LSEQ * 4);  // x_proj out (padded 40->48)
    float* dlt    = (float*)alloc((size_t)BATCH * DI * LSEQ * 4);
    bf16*  ybf    = (bf16*) alloc((size_t)BL * DI * 2);            // scan out [b][l][d]
    float* scale1 = (float*)alloc(CDIM * 4);
    float* shift1 = (float*)alloc(CDIM * 4);
    float* scale2 = (float*)alloc(CDIM * 4);
    float* shift2 = (float*)alloc(CDIM * 4);
    bf16*  nin_pk   = (bf16*)alloc((size_t)8  * 4 * 512 * 2);   // M=128 K=128
    bf16*  nin2_pk  = (bf16*)alloc((size_t)8  * 4 * 512 * 2);
    bf16*  inpj_pk0 = (bf16*)alloc((size_t)32 * 4 * 512 * 2);   // M=512 K=128
    bf16*  inpj_pk1 = (bf16*)alloc((size_t)32 * 4 * 512 * 2);   // channel-flipped
    bf16*  xpj_pk   = (bf16*)alloc((size_t)3  * 8 * 512 * 2);   // M=48(pad) K=256
    bf16*  outpj_pk = (bf16*)alloc((size_t)8  * 8 * 512 * 2);   // M=128 K=256
    bf16*  sb16     = xb16;  // reuse: x transpose no longer needed after nin GEMM
    float* pre2     = pre1;  // reuse

    hipMemsetAsync(xmamba, 0, (size_t)BL * CDIM * 4, stream);

    // ----- weight packing (tiny) -----
    pack_a_kernel<<<blk256(8  * 4 * 512), 256, 0, stream>>>(nin_w,     nin_pk,   128, 8,  128, 0);
    pack_a_kernel<<<blk256(8  * 4 * 512), 256, 0, stream>>>(nin2_w,    nin2_pk,  128, 8,  128, 0);
    pack_a_kernel<<<blk256(32 * 4 * 512), 256, 0, stream>>>(in_proj_w, inpj_pk0, 512, 32, 128, 0);
    pack_a_kernel<<<blk256(32 * 4 * 512), 256, 0, stream>>>(in_proj_w, inpj_pk1, 512, 32, 128, 1);
    pack_a_kernel<<<blk256(3  * 8 * 512), 256, 0, stream>>>(x_proj_w,  xpj_pk,   40,  3,  256, 0);
    pack_a_kernel<<<blk256(8  * 8 * 512), 256, 0, stream>>>(out_proj_w,outpj_pk, 128, 8,  256, 0);

    const size_t nBCL = (size_t)BATCH * CDIM * LSEQ;   // 2M
    const size_t nBDL = (size_t)BATCH * DI * LSEQ;     // 4M
    const dim3 gemmBlk(128);

    // ----- nin (1x1 conv) + BN1 + ReLU -----
    transpose_x_kernel<<<blk256(nBCL), 256, 0, stream>>>(x, xb16);
    wmma_gemm_kernel<<<dim3(1024, 2), gemmBlk, 0, stream>>>(nin_pk, xb16, pre1, 8, 4, 128, 0);
    bn_stats_kernel<<<128, 256, 0, stream>>>(pre1, gamma, beta, scale1, shift1);
    bn_relu_t_kernel<<<blk256(nBCL), 256, 0, stream>>>(pre1, scale1, shift1, xn_f32, xn_b16);

    // ----- 4-direction mamba -----
    for (int dir = 0; dir < 4; ++dir) {
        int fc = dir & 1;          // channel flip (reference axis 2)
        int fl = (dir >> 1) & 1;   // sequence flip (reference axis 1)
        const bf16* inpk = fc ? inpj_pk1 : inpj_pk0;

        wmma_gemm_kernel<<<dim3(1024, 8), gemmBlk, 0, stream>>>(inpk, xn_b16, xz, 32, 4, 512, fl);
        conv_silu_kernel<<<blk256(nBDL), 256, 0, stream>>>(xz, conv_w, conv_b, xca, xcab);
        wmma_gemm_kernel<<<dim3(1024, 1), gemmBlk, 0, stream>>>(xpj_pk, xcab, xdbl, 3, 8, 48, 0);
        dt_softplus_kernel<<<blk256(nBDL), 256, 0, stream>>>(xdbl, dt_proj_w, dt_proj_b, dlt);
        ssm_scan_kernel<<<BATCH * 128, 32, 0, stream>>>(dlt, xca, xdbl, xz, A_log, Dp, ybf);
        wmma_outproj_kernel<<<dim3(1024, 2), gemmBlk, 0, stream>>>(outpj_pk, ybf, xmamba, fl, fc);
    }

    // ----- residual + nin2 + BN2 + ReLU -----
    add_avg_kernel<<<blk256((size_t)BL * CDIM), 256, 0, stream>>>(xmamba, xn_f32, sb16);
    wmma_gemm_kernel<<<dim3(1024, 2), gemmBlk, 0, stream>>>(nin2_pk, sb16, pre2, 8, 4, 128, 0);
    bn_stats_kernel<<<128, 256, 0, stream>>>(pre2, gamma2, beta2, scale2, shift2);
    bn_relu_out_kernel<<<blk256(nBCL), 256, 0, stream>>>(pre2, scale2, shift2, out);
}
